// AutoRNN_63376537419776
// MI455X (gfx1250) — matrix operand order repaired and checked by reading, not compiled, not run on hardware
//
#include <hip/hip_runtime.h>
#include <cstdint>

#define T_SEQ 32768
#define D_IN  1024
#define H_DIM 1024
#define O_DIM 256
#define SCAN_BLOCKS 16
#define COLS_PER_BLOCK 64   // H_DIM / SCAN_BLOCKS
#define KCHUNKS (D_IN / 32) // 32

typedef __bf16 bf16_t;
typedef bf16_t v16bf __attribute__((ext_vector_type(16)));
typedef float  v8f   __attribute__((ext_vector_type(8)));
typedef unsigned int v4u __attribute__((ext_vector_type(4)));
typedef int    v4i   __attribute__((ext_vector_type(4)));
typedef int    v8i   __attribute__((ext_vector_type(8)));

// Split an fp32 value into hi/lo bf16 parts (round-to-nearest-even) such that
// hi + lo ~= f to ~fp32 accuracy when products are accumulated in fp32.
__device__ __forceinline__ void split2bf(float f, bf16_t& hi, bf16_t& lo) {
  unsigned u  = __float_as_uint(f);
  unsigned short hb = (unsigned short)((u + 0x7FFFu + ((u >> 16) & 1u)) >> 16);
  float hf = __uint_as_float((unsigned)hb << 16);
  float lf = f - hf;
  unsigned ul = __float_as_uint(lf);
  unsigned short lb = (unsigned short)((ul + 0x7FFFu + ((ul >> 16) & 1u)) >> 16);
  hi = __builtin_bit_cast(bf16_t, hb);
  lo = __builtin_bit_cast(bf16_t, lb);
}

// ---------------------------------------------------------------------------
// Prep A: X[T,D] fp32 -> Ahi/Alo bf16 in WMMA A-fragment order.
// Fragment (tileRow, kChunk): lane l<16 holds row tileRow*16+l,
// K = {kc*32+0..7, kc*32+16..23}; lane>=16 holds K = {+8..15, +24..31}.
// Linear: (((tileRow*KCHUNKS + kChunk)*32 + lane)*16 + e). One wave per frag.
// ---------------------------------------------------------------------------
__global__ __launch_bounds__(256) void prep_a(
    const float* __restrict__ X, bf16_t* __restrict__ Ahi, bf16_t* __restrict__ Alo) {
  const int lane = threadIdx.x & 31;
  const int wv   = threadIdx.x >> 5;
  const int frag = blockIdx.x * 8 + wv;      // T/16 * KCHUNKS = 65536 frags
  const int kc   = frag & (KCHUNKS - 1);
  const int tr   = frag >> 5;
  const int laneLo = lane & 15, laneHi = lane >> 4;
  const float* ap = X + (size_t)(tr * 16 + laneLo) * D_IN + kc * 32 + laneHi * 8;
  bf16_t* oh = Ahi + ((size_t)frag * 32 + lane) * 16;
  bf16_t* ol = Alo + ((size_t)frag * 32 + lane) * 16;
#pragma unroll
  for (int e = 0; e < 8; ++e) {
    split2bf(ap[e],      oh[e],     ol[e]);
    split2bf(ap[e + 16], oh[e + 8], ol[e + 8]);
  }
}

// ---------------------------------------------------------------------------
// Prep B: Wx[D,H] fp32 -> Bhi/Blo bf16 in WMMA B-fragment order.
// Fragment (colTile, kChunk): lane l<16 holds col colTile*16+l,
// K = {kc*32+0..7, kc*32+16..23}; lane>=16 holds K = {+8..15, +24..31}. Linear like prep_a. One wave per fragment.
// ---------------------------------------------------------------------------
__global__ __launch_bounds__(256) void prep_b(
    const float* __restrict__ Wx, bf16_t* __restrict__ Bhi, bf16_t* __restrict__ Blo) {
  const int lane = threadIdx.x & 31;
  const int wv   = threadIdx.x >> 5;
  const int frag = blockIdx.x * 8 + wv;      // H/16 * KCHUNKS = 2048 frags
  const int kc   = frag & (KCHUNKS - 1);
  const int ct   = frag >> 5;
  const int laneLo = lane & 15, laneHi = lane >> 4;
  const float* bp = Wx + (size_t)(kc * 32 + laneHi * 8) * H_DIM + ct * 16 + laneLo;
  bf16_t* oh = Bhi + ((size_t)frag * 32 + lane) * 16;
  bf16_t* ol = Blo + ((size_t)frag * 32 + lane) * 16;
#pragma unroll
  for (int e = 0; e < 16; ++e)
    split2bf(bp[(size_t)(e + (e & 8)) * H_DIM], oh[e], ol[e]);
}

// ---------------------------------------------------------------------------
// Phase 1: Xi = X @ Wx via bf16 WMMA (hi/lo split: 3 WMMAs per tile pair).
// Block = 256 threads (8 waves). Wave owns a 16x128 strip (8 f32 accumulators).
// All operand loads are contiguous v16bf (2x global_load_b128 each).
// ---------------------------------------------------------------------------
__global__ __launch_bounds__(256) void xw_gemm_wmma(
    const bf16_t* __restrict__ Ahi, const bf16_t* __restrict__ Alo,
    const bf16_t* __restrict__ Bhi, const bf16_t* __restrict__ Blo,
    float* __restrict__ Xi) {
  const int lane   = threadIdx.x & 31;
  const int wave   = threadIdx.x >> 5;
  const int laneLo = lane & 15;
  const int laneHi = lane >> 4;
  const int rowTile  = blockIdx.x * 8 + wave;  // 16 rows per tile
  const int colTile0 = blockIdx.y * 8;         // 8 col tiles = 128 cols

  const v16bf* ahp = (const v16bf*)Ahi + (size_t)rowTile * KCHUNKS * 32 + lane;
  const v16bf* alp = (const v16bf*)Alo + (size_t)rowTile * KCHUNKS * 32 + lane;
  const v16bf* bhp = (const v16bf*)Bhi + (size_t)colTile0 * KCHUNKS * 32 + lane;
  const v16bf* blp = (const v16bf*)Blo + (size_t)colTile0 * KCHUNKS * 32 + lane;

  v8f acc[8];
#pragma unroll
  for (int t = 0; t < 8; ++t) acc[t] = (v8f){0.f,0.f,0.f,0.f,0.f,0.f,0.f,0.f};

  for (int kc = 0; kc < KCHUNKS; ++kc) {
    const v16bf ah = ahp[(size_t)kc * 32];
    const v16bf al = alp[(size_t)kc * 32];
#pragma unroll
    for (int t = 0; t < 8; ++t) {
      const v16bf bh = bhp[((size_t)t * KCHUNKS + kc) * 32];
      const v16bf bl = blp[((size_t)t * KCHUNKS + kc) * 32];
      acc[t] = __builtin_amdgcn_wmma_f32_16x16x32_bf16(false, ah, false, bh,
                                                       (short)0, acc[t], false, false);
      acc[t] = __builtin_amdgcn_wmma_f32_16x16x32_bf16(false, ah, false, bl,
                                                       (short)0, acc[t], false, false);
      acc[t] = __builtin_amdgcn_wmma_f32_16x16x32_bf16(false, al, false, bh,
                                                       (short)0, acc[t], false, false);
    }
  }
  // C/D layout: VGPR r, lanes 0-15 -> M=r, lanes 16-31 -> M=r+8; N=laneLo.
  const int m0 = rowTile * 16 + laneHi * 8;
#pragma unroll
  for (int t = 0; t < 8; ++t) {
    const int n = (colTile0 + t) * 16 + laneLo;
#pragma unroll
    for (int r = 0; r < 8; ++r)
      Xi[(size_t)(m0 + r) * H_DIM + n] = acc[t][r];
  }
}

// ---------------------------------------------------------------------------
// Phase 2: sequential recurrence. 16 persistent blocks, block g owns Wh columns
// [g*64, g*64+64) resident in LDS (256 KB, loaded once by the Tensor Data
// Mover). Per step: 1024 threads (col = tid&63, part = tid>>6) compute 64-long
// fp32 partial dots, deterministic LDS tree-reduce, tanh, publish 64 h values
// to a global ping-pong buffer, then a monotonic atomic-counter grid barrier.
// ---------------------------------------------------------------------------
__global__ __launch_bounds__(1024) void rnn_scan(
    const float* __restrict__ Xi, const float* __restrict__ Wh,
    const float* __restrict__ bh, float* __restrict__ hbuf,
    unsigned* __restrict__ cnt) {
  extern __shared__ float whLds[];      // [1024][64] fp32 column slice, 256 KB
  __shared__ float hLds[H_DIM];         // current hidden state (full)
  __shared__ float red[1024];           // [part][col] partials

  const int tid     = threadIdx.x;
  const int g       = blockIdx.x;
  const int colBase = g * COLS_PER_BLOCK;

  // --- TDM: load Wh[:, colBase:colBase+64) into LDS as a 2D tile ------------
  if (tid < 32) {                       // one wave issues the DMA (EXEC ignored)
    const uint32_t lds_addr = (uint32_t)(uintptr_t)(&whLds[0]);   // low 32 bits = LDS byte offset
    const uint64_t ga = (uint64_t)(uintptr_t)(Wh + colBase);      // tile start address
    v4u g0;
    g0[0] = 1u;                                            // count=1, user descriptor
    g0[1] = lds_addr;                                      // lds_addr
    g0[2] = (uint32_t)ga;                                  // global_addr[31:0]
    g0[3] = (uint32_t)((ga >> 32) & 0x1FFFFFFu) | (2u << 30); // addr[56:32] | type=2
    v8i g1;
    g1[0] = (int)(2u << 16);                               // wg_mask=0, data_size=4B
    g1[1] = (int)((uint32_t)H_DIM << 16);                  // tensor_dim0 = 1024 (lo16)
    g1[2] = (int)((uint32_t)H_DIM << 16);                  // tdim0 hi=0 | tensor_dim1 lo16
    g1[3] = (int)((uint32_t)COLS_PER_BLOCK << 16);         // tdim1 hi=0 | tile_dim0 = 64
    g1[4] = (int)(uint32_t)H_DIM;                          // tile_dim1 = 1024, tile_dim2 = 0
    g1[5] = (int)H_DIM;                                    // tensor_dim0_stride = 1024 (lo32)
    g1[6] = 0;                                             // stride0 hi | stride1 lo16 = 0
    g1[7] = (int)((uint32_t)((uint64_t)H_DIM * H_DIM >> 16)); // stride1[47:16]
    v4i g2 = (v4i){0, 0, 0, 0};
    v4i g3 = (v4i){0, 0, 0, 0};
#if defined(__clang_major__) && (__clang_major__ >= 23)
    v8i gx = (v8i){0, 0, 0, 0, 0, 0, 0, 0};
    __builtin_amdgcn_tensor_load_to_lds(g0, g1, g2, g3, gx, 0);
#else
    __builtin_amdgcn_tensor_load_to_lds(g0, g1, g2, g3, 0);
#endif
    __builtin_amdgcn_s_wait_tensorcnt((short)0);
  }
  hLds[tid] = Xi[tid];                  // h0 = Xi[0] (reference quirk: no tanh/bias)
  __syncthreads();

  const int col   = tid & 63;
  const int part  = tid >> 6;           // 0..15
  const int ibase = part * 64;

  for (int t = 1; t < T_SEQ; ++t) {
    float partial = 0.f;
#pragma unroll 8
    for (int i = 0; i < 64; ++i)
      partial = fmaf(hLds[ibase + i], whLds[(ibase + i) * 64 + col], partial);
    red[part * 64 + col] = partial;     // conflict-free: lanes hit consecutive banks
    __syncthreads();

    float* hnext = hbuf + (size_t)(t & 1) * H_DIM;
    if (tid < COLS_PER_BLOCK) {
      float s = Xi[(size_t)t * H_DIM + colBase + tid] + bh[colBase + tid];
#pragma unroll
      for (int p = 0; p < 16; ++p) s += red[p * 64 + tid];  // deterministic reduce
      hnext[colBase + tid] = tanhf(s);
      __threadfence();                  // make h visible at device scope
    }
    __syncthreads();

    if (tid == 0) {                     // monotonic-counter grid barrier
      __hip_atomic_fetch_add(cnt, 1u, __ATOMIC_ACQ_REL, __HIP_MEMORY_SCOPE_AGENT);
      const unsigned target = (unsigned)SCAN_BLOCKS * (unsigned)t;
      while (__hip_atomic_load(cnt, __ATOMIC_ACQUIRE, __HIP_MEMORY_SCOPE_AGENT) < target)
        __builtin_amdgcn_s_sleep(1);
    }
    __syncthreads();
    hLds[tid] = hnext[tid];             // pull the full new h into LDS
    if (t + 1 < T_SEQ)                  // warm L2 for next step's xi slice
      __builtin_prefetch(&Xi[(size_t)(t + 1) * H_DIM + colBase + (tid & 63)], 0, 0);
    __syncthreads();
  }
}

// ---------------------------------------------------------------------------
// Phase 3: logit[O] = hT @ Wy + by. One 256-thread block, coalesced Wy reads.
// ---------------------------------------------------------------------------
__global__ __launch_bounds__(256) void out_proj(
    const float* __restrict__ h, const float* __restrict__ Wy,
    const float* __restrict__ by, float* __restrict__ out) {
  const int j = threadIdx.x;
  float s = by[j];
  for (int i = 0; i < H_DIM; ++i)
    s = fmaf(h[i], Wy[(size_t)i * O_DIM + j], s);
  out[j] = s;
}

extern "C" void kernel_launch(void* const* d_in, const int* in_sizes, int n_in,
                              void* d_out, int out_size, void* d_ws, size_t ws_size,
                              hipStream_t stream) {
  (void)in_sizes; (void)n_in; (void)out_size; (void)ws_size;
  const float* X  = (const float*)d_in[0];
  const float* Wx = (const float*)d_in[1];
  const float* Wh = (const float*)d_in[2];
  const float* Wy = (const float*)d_in[3];
  const float* bh = (const float*)d_in[4];
  const float* by = (const float*)d_in[5];
  float* out = (float*)d_out;

  // Workspace layout (bytes):
  //   Xi  fp32 [T,H]          128 MiB
  //   Ahi/Alo bf16 [T,D]       64 MiB each (fragment-swizzled)
  //   Bhi/Blo bf16 [D,H]        2 MiB each (fragment-swizzled)
  //   h ping-pong [2,H] fp32, barrier counter
  char* ws = (char*)d_ws;
  float*  Xi   = (float*)ws;
  bf16_t* Ahi  = (bf16_t*)(ws + (size_t)128 * 1024 * 1024);
  bf16_t* Alo  = (bf16_t*)(ws + (size_t)192 * 1024 * 1024);
  bf16_t* Bhi  = (bf16_t*)(ws + (size_t)256 * 1024 * 1024);
  bf16_t* Blo  = (bf16_t*)(ws + (size_t)258 * 1024 * 1024);
  float*  hbuf = (float*)(ws + (size_t)260 * 1024 * 1024);
  unsigned* cnt = (unsigned*)(hbuf + 2 * H_DIM);

  hipMemsetAsync(cnt, 0, sizeof(unsigned), stream);   // deterministic barrier state

  // fp32 -> split-bf16, fragment-swizzled (memory-bound one-pass)
  prep_a<<<dim3((T_SEQ / 16) * KCHUNKS / 8), dim3(256), 0, stream>>>(X, Ahi, Alo);
  prep_b<<<dim3((H_DIM / 16) * KCHUNKS / 8), dim3(256), 0, stream>>>(Wx, Bhi, Blo);

  xw_gemm_wmma<<<dim3(T_SEQ / 128, H_DIM / 128), dim3(256), 0, stream>>>(
      Ahi, Alo, Bhi, Blo, Xi);

  rnn_scan<<<dim3(SCAN_BLOCKS), dim3(1024),
             H_DIM * COLS_PER_BLOCK * sizeof(float), stream>>>(Xi, Wh, bh, hbuf, cnt);

  const float* hT = hbuf + (size_t)((T_SEQ - 1) & 1) * H_DIM;
  out_proj<<<dim3(1), dim3(256), 0, stream>>>(hT, Wy, by, out);
}
